// TextPerformer_51084341018972
// MI455X (gfx1250) — matrix operand order repaired and checked by reading, not compiled, not run on hardware
//
#include <hip/hip_runtime.h>
#include <hip/hip_bf16.h>
#include <math.h>

typedef _Float16 half_t;
typedef __attribute__((ext_vector_type(16))) _Float16 v16h;
typedef __attribute__((ext_vector_type(8)))  _Float16 v8h;
typedef __attribute__((ext_vector_type(8)))  float    v8f;

// ---- problem constants (from reference) ----
constexpr int B_   = 8;
constexpr int N_   = 2048;
constexpr int C_   = 512;
constexpr int H_   = 8;
constexpr int L_   = 8;
constexpr int M_   = 128;
constexpr int NC_  = 2;
constexpr int HD_  = C_ / H_;        // 64
constexpr int HID_ = 4 * C_;         // 2048
constexpr int NT_  = N_ + 1;         // 2049
constexpr int NTP_ = 2080;           // NT_ padded to %32 (pads zero-filled once)
constexpr int R_   = B_ * NT_;       // 16392
constexpr int HM_  = H_ * M_;        // 1024
constexpr float SCALE_    = 0.125f;
constexpr float LN_EPS_   = 1e-5f;
constexpr float FEAT_EPS_ = 1e-6f;

#ifndef USE_ASYNC_LDS
#define USE_ASYNC_LDS 1
#endif

// =====================================================================
// CDNA5 async global->LDS staging (ASYNCcnt-tracked, bypasses VGPRs)
// =====================================================================
__device__ inline void async_ld_b128(const void* g, void* l) {
#if USE_ASYNC_LDS
  unsigned loff = (unsigned)(size_t)l;   // generic LDS addr truncates to offset
  asm volatile("global_load_async_to_lds_b128 %0, %1, off"
               : : "v"(loff), "v"(g) : "memory");
#else
  *(v8h*)l = *(const v8h*)g;
#endif
}
__device__ inline void async_wait0() {
#if USE_ASYNC_LDS
#if __has_builtin(__builtin_amdgcn_s_wait_asynccnt)
  __builtin_amdgcn_s_wait_asynccnt(0);
#else
  asm volatile("s_wait_asynccnt 0x0" : : : "memory");
#endif
#endif
}
// allow the just-issued prefetch group (<=4 per-thread loads) to stay in flight
__device__ inline void async_wait4() {
#if USE_ASYNC_LDS
#if __has_builtin(__builtin_amdgcn_s_wait_asynccnt)
  __builtin_amdgcn_s_wait_asynccnt(4);
#else
  asm volatile("s_wait_asynccnt 0x4" : : : "memory");
#endif
#endif
}

// =====================================================================
// WMMA helpers (CDNA5 v_wmma_f32_16x16x32_f16, wave32)
// =====================================================================
__device__ inline v8f wmma16(v16h a, v16h b, v8f c) {
  return __builtin_amdgcn_wmma_f32_16x16x32_f16(
      false, a, false, b, (short)0, c, false, false);
}

__device__ inline v16h frag_a(const half_t* t, int ld, int lane) {
  int row = lane & 15;
  int kb  = (lane >> 4) ? 8 : 0;
  v8h lo = *(const v8h*)&t[row * ld + kb];
  v8h hi = *(const v8h*)&t[row * ld + kb + 16];
  v16h a;
#pragma unroll
  for (int i = 0; i < 8; ++i) { a[i] = lo[i]; a[8 + i] = hi[i]; }
  return a;
}

__device__ inline v16h frag_bt(const half_t* tT, int ld, int lane) {
  int col = lane & 15;
  int kb  = (lane >> 4) ? 8 : 0;
  v8h lo = *(const v8h*)&tT[col * ld + kb];
  v8h hi = *(const v8h*)&tT[col * ld + kb + 16];
  v16h b;
#pragma unroll
  for (int i = 0; i < 8; ++i) { b[i] = lo[i]; b[8 + i] = hi[i]; }
  return b;
}

// =====================================================================
// Transposing fp32 -> fp16 weight convert: src (K x N) -> dst (N x K), batched.
// =====================================================================
__global__ void k_convert_T(const float* __restrict__ src,
                            half_t* __restrict__ dst, int K, int N) {
  __shared__ float tile[32][33];
  long base = (long)blockIdx.z * K * N;
  int k0 = blockIdx.x * 32, n0 = blockIdx.y * 32;
  int tx = threadIdx.x & 31, ty = threadIdx.x >> 5;
#pragma unroll
  for (int i = 0; i < 32; i += 8)
    tile[ty + i][tx] = src[base + (long)(k0 + ty + i) * N + n0 + tx];
  __syncthreads();
#pragma unroll
  for (int i = 0; i < 32; i += 8)
    dst[base + (long)(n0 + ty + i) * K + k0 + tx] = (half_t)tile[tx][ty + i];
}

__global__ void k_zero_h(half_t* __restrict__ p, long n) {
  long i = (long)blockIdx.x * blockDim.x + threadIdx.x;
  long st = (long)gridDim.x * blockDim.x;
  for (; i < n; i += st) p[i] = (half_t)0.f;
}

__global__ void k_embed(const int* __restrict__ ids,
                        const float* __restrict__ tok,
                        const float* __restrict__ cls,
                        const float* __restrict__ pos,
                        float* __restrict__ X) {
  int row = blockIdx.x;
  int b = row / NT_, t = row % NT_;
  for (int c = threadIdx.x; c < C_; c += blockDim.x) {
    float e = (t == 0) ? cls[c] : tok[(long)ids[b * N_ + t - 1] * C_ + c];
    X[(long)row * C_ + c] = e + pos[t * C_ + c];
  }
}

__global__ void k_layernorm(const float* __restrict__ X,
                            const float* __restrict__ g,
                            const float* __restrict__ beta,
                            half_t* __restrict__ Y) {
  __shared__ float red[256];
  int row = blockIdx.x, tid = threadIdx.x;
  const float* xr = X + (long)row * C_;
  float v0 = xr[tid], v1 = xr[tid + 256];
  red[tid] = v0 + v1;
  __syncthreads();
  for (int s = 128; s > 0; s >>= 1) { if (tid < s) red[tid] += red[tid + s]; __syncthreads(); }
  float mu = red[0] / (float)C_;
  __syncthreads();
  float d0 = v0 - mu, d1 = v1 - mu;
  red[tid] = d0 * d0 + d1 * d1;
  __syncthreads();
  for (int s = 128; s > 0; s >>= 1) { if (tid < s) red[tid] += red[tid + s]; __syncthreads(); }
  float rstd = rsqrtf(red[0] / (float)C_ + LN_EPS_);
  half_t* yr = Y + (long)row * C_;
  yr[tid]       = (half_t)(d0 * rstd * g[tid] + beta[tid]);
  yr[tid + 256] = (half_t)(d1 * rstd * g[tid + 256] + beta[tid + 256]);
}

// =====================================================================
// Generic WMMA GEMM with async double-buffered staging.
// out = A(R x K) @ W(K x N) + bias, W TRANSPOSED (N x K).
// block 256 (8 waves); tile 128x128; wave 16x128; K%32==0, N%128==0.
//   EPI 0: fp16 *scale | 1: fp16 *mask | 2: fp32 += | 3: GELU fp16
//   EPI 4: fp16 *mask, store transposed to Vt[b,h,d,NTP]
// =====================================================================
template <int EPI>
__global__ void k_gemm(const half_t* __restrict__ A, int lda,
                       const half_t* __restrict__ Wt,
                       const float* __restrict__ bias,
                       float* __restrict__ outF, half_t* __restrict__ outH,
                       int ldo, int Rrows, int K, float scale,
                       const int* __restrict__ amask) {
  __shared__ half_t sA[2][128][40];
  __shared__ half_t sBt[2][128][40];
  int tid = threadIdx.x;
  int wave = tid >> 5, lane = tid & 31;
  int row0 = blockIdx.x * 128;
  int col0 = blockIdx.y * 128;
  int wrow = row0 + wave * 16;

  int r0 = tid >> 2, r1 = r0 + 64;
  int seg = (tid & 3) * 8;
  int gr0 = row0 + r0; if (gr0 >= Rrows) gr0 = Rrows - 1;  // clamped rows are
  int gr1 = row0 + r1; if (gr1 >= Rrows) gr1 = Rrows - 1;  // store-guarded

  auto issue = [&](int buf, int k0) {
    async_ld_b128(&A[(long)gr0 * lda + k0 + seg], &sA[buf][r0][seg]);
    async_ld_b128(&A[(long)gr1 * lda + k0 + seg], &sA[buf][r1][seg]);
    async_ld_b128(&Wt[(long)(col0 + r0) * K + k0 + seg], &sBt[buf][r0][seg]);
    async_ld_b128(&Wt[(long)(col0 + r1) * K + k0 + seg], &sBt[buf][r1][seg]);
  };

  v8f acc[8] = {};
  int nk = K >> 5;
  issue(0, 0);
  for (int ki = 0; ki < nk; ++ki) {
    int buf = ki & 1;
    if (ki + 1 < nk) { issue(buf ^ 1, (ki + 1) * 32); async_wait4(); }
    else             { async_wait0(); }
    __syncthreads();
    v16h af = frag_a(&sA[buf][wave * 16][0], 40, lane);
#pragma unroll
    for (int t = 0; t < 8; ++t) {
      v16h bf = frag_bt(&sBt[buf][t * 16][0], 40, lane);
      acc[t] = wmma16(af, bf, acc[t]);
    }
    __syncthreads();
  }

  int nn = lane & 15;
  int mb = (lane >> 4) * 8;
#pragma unroll
  for (int t = 0; t < 8; ++t) {
    int col = col0 + t * 16 + nn;
    float bv = bias ? bias[col] : 0.f;
#pragma unroll
    for (int j = 0; j < 8; ++j) {
      int row = wrow + mb + j;
      if (row >= Rrows) continue;
      float v = acc[t][j] + bv;
      if constexpr (EPI == 0) {
        outH[(long)row * ldo + col] = (half_t)(v * scale);
      } else if constexpr (EPI == 1) {
        int b = row / NT_, tk = row % NT_;
        float mk = (tk == 0) ? 1.f : (float)amask[b * N_ + tk - 1];
        outH[(long)row * ldo + col] = (half_t)(v * mk);
      } else if constexpr (EPI == 2) {
        outF[(long)row * ldo + col] += v;
      } else if constexpr (EPI == 3) {
        float gl = 0.5f * v * (1.f + erff(v * 0.70710678118f));
        outH[(long)row * ldo + col] = (half_t)gl;
      } else {  // EPI 4: masked, transposed store -> Vt[b, h, d, tk]
        int b = row / NT_, tk = row % NT_;
        float mk = (tk == 0) ? 1.f : (float)amask[b * N_ + tk - 1];
        int hh = col >> 6, d = col & 63;
        outH[((long)(b * H_ + hh) * HD_ + d) * NTP_ + tk] = (half_t)(v * mk);
      }
    }
  }
}

// =====================================================================
// Performer feature map: F = exp(q·P - 0.5*||q||^2) + eps
// TR=false: F is R x HM (for qf). TR=true: F is [b,h,m,NTP] (for kf).
// =====================================================================
template <bool TR>
__global__ void k_feature(const half_t* __restrict__ Q,
                          const half_t* __restrict__ Pt,   // H x M x HD
                          half_t* __restrict__ F) {
  __shared__ half_t sQ[32][72];
  __shared__ half_t sPt[128][72];
  __shared__ float  sF[32][128];
  __shared__ float  sXsq[32];
  int tid = threadIdx.x;
  int wave = tid >> 5, lane = tid & 31;
  int h = blockIdx.y;
  int row0 = blockIdx.x * 32;

  {
    int r = tid >> 3, seg = (tid & 7) * 8;
    int gr = row0 + r; if (gr >= R_) gr = R_ - 1;
    async_ld_b128(&Q[(long)gr * C_ + h * HD_ + seg], &sQ[r][seg]);
  }
  {
    const half_t* pb = Pt + (long)h * M_ * HD_;
#pragma unroll
    for (int it = 0; it < 4; ++it) {
      int j = tid + it * 256;
      int r = j >> 3, seg = (j & 7) * 8;
      async_ld_b128(&pb[(long)r * HD_ + seg], &sPt[r][seg]);
    }
  }
  async_wait0();
  __syncthreads();

  int rt = wave >> 2, cq = wave & 3;
  v8f acc[2] = {};
#pragma unroll
  for (int k = 0; k < 2; ++k) {
    v16h af = frag_a(&sQ[rt * 16][k * 32], 72, lane);
#pragma unroll
    for (int t = 0; t < 2; ++t) {
      v16h bf = frag_bt(&sPt[cq * 32 + t * 16][k * 32], 72, lane);
      acc[t] = wmma16(af, bf, acc[t]);
    }
  }
  int nn = lane & 15, mb = (lane >> 4) * 8;
#pragma unroll
  for (int t = 0; t < 2; ++t)
#pragma unroll
    for (int j = 0; j < 8; ++j)
      sF[rt * 16 + mb + j][cq * 32 + t * 16 + nn] = acc[t][j];
  __syncthreads();

  if (tid < 32) {
    float s = 0.f;
#pragma unroll
    for (int c = 0; c < HD_; ++c) {
      float q = (float)sQ[tid][c];
      s += q * q;
    }
    sXsq[tid] = 0.5f * s;
  }
  __syncthreads();

  for (int i = tid; i < 32 * 128; i += 256) {
    int r = i >> 7, c = i & 127;
    int gr = row0 + r;
    if (gr < R_) {
      float val = expf(sF[r][c] - sXsq[r]) + FEAT_EPS_;
      if constexpr (TR) {
        int b = gr / NT_, tk = gr - b * NT_;
        F[((long)(b * H_ + h) * M_ + c) * NTP_ + tk] = (half_t)val;
      } else {
        F[(long)gr * HM_ + h * M_ + c] = (half_t)val;
      }
    }
  }
}

// =====================================================================
// kv^T[b,h]: kv = kf^T @ v (M x HD), stored transposed (HD x M).
// Inputs already transposed (KFt [bh,m,NTP], Vt [bh,d,NTP]; pads are zero)
// -> fully async, double-buffered, vectorized staging; no guards.
// =====================================================================
__global__ void k_kv(const half_t* __restrict__ KFt,
                     const half_t* __restrict__ Vt,
                     half_t* __restrict__ KVt) {
  __shared__ half_t sKt[2][128][40];
  __shared__ half_t sVt[2][64][40];
  int tid = threadIdx.x;
  int wave = tid >> 5, lane = tid & 31;
  int bh = blockIdx.x;
  int m0 = wave * 16;

  int r0 = tid >> 2, r1 = r0 + 64;
  int seg = (tid & 3) * 8;

  auto issue = [&](int buf, int n0) {
    async_ld_b128(&KFt[((long)bh * M_ + r0) * NTP_ + n0 + seg], &sKt[buf][r0][seg]);
    async_ld_b128(&KFt[((long)bh * M_ + r1) * NTP_ + n0 + seg], &sKt[buf][r1][seg]);
    async_ld_b128(&Vt[((long)bh * HD_ + r0) * NTP_ + n0 + seg], &sVt[buf][r0][seg]);
  };

  v8f acc[4] = {};
  constexpr int nk = NTP_ / 32;  // 65
  issue(0, 0);
  for (int ki = 0; ki < nk; ++ki) {
    int buf = ki & 1;
    if (ki + 1 < nk) { issue(buf ^ 1, (ki + 1) * 32); async_wait4(); }
    else             { async_wait0(); }
    __syncthreads();
    v16h af = frag_a(&sKt[buf][m0][0], 40, lane);
#pragma unroll
    for (int t = 0; t < 4; ++t) {
      v16h bf = frag_bt(&sVt[buf][t * 16][0], 40, lane);
      acc[t] = wmma16(af, bf, acc[t]);
    }
    __syncthreads();
  }
  int nn = lane & 15, mb = (lane >> 4) * 8;
#pragma unroll
  for (int t = 0; t < 4; ++t)
#pragma unroll
    for (int j = 0; j < 8; ++j) {
      int m = m0 + mb + j, d = t * 16 + nn;
      KVt[((long)bh * HD_ + d) * M_ + m] = (half_t)acc[t][j];
    }
}

// ksum[bh,m] = sum_n KFt[bh,m,n]
__global__ void k_ksum(const half_t* __restrict__ KFt, float* __restrict__ KS) {
  int gidx = blockIdx.x * 256 + threadIdx.x;    // 64*128 threads
  int bh = gidx >> 7, m = gidx & 127;
  const half_t* p = &KFt[((long)bh * M_ + m) * NTP_];
  float s = 0.f;
  for (int n = 0; n < NT_; ++n) s += (float)p[n];
  KS[(long)bh * M_ + m] = s;
}

__global__ void k_denom(const half_t* __restrict__ QF,
                        const float* __restrict__ KS,
                        float* __restrict__ D) {
  __shared__ float red[128];
  int row = blockIdx.x, h = blockIdx.y, m = threadIdx.x;
  int b = row / NT_;
  red[m] = (float)QF[(long)row * HM_ + h * M_ + m] * KS[((long)b * H_ + h) * M_ + m];
  __syncthreads();
  for (int s = 64; s > 0; s >>= 1) { if (m < s) red[m] += red[m + s]; __syncthreads(); }
  if (m == 0) D[(long)row * H_ + h] = fmaxf(red[0], 1e-6f);
}

// ctx = (qf @ kv) / denom -> CTX (R x C) fp16
__global__ void k_ctx(const half_t* __restrict__ QF,
                      const half_t* __restrict__ KVt,
                      const float* __restrict__ D,
                      half_t* __restrict__ CTX) {
  __shared__ half_t sA[64][40];
  __shared__ half_t sBt[64][40];
  int tid = threadIdx.x;
  int wave = tid >> 5, lane = tid & 31;
  int bh = blockIdx.y;
  int b = bh >> 3, h = bh & 7;
  int rt = wave >> 1;
  int ch = (wave & 1) * 32;
  int tok0 = blockIdx.x * 64;

  int ar = tid >> 2, aseg = (tid & 3) * 8;
  int tokA = tok0 + ar; if (tokA >= NT_) tokA = NT_ - 1;

  v8f acc[2] = {};
  for (int k0 = 0; k0 < M_; k0 += 32) {
    async_ld_b128(&QF[(long)(b * NT_ + tokA) * HM_ + h * M_ + k0 + aseg],
                  &sA[ar][aseg]);
    async_ld_b128(&KVt[((long)bh * HD_ + ar) * M_ + k0 + aseg],
                  &sBt[ar][aseg]);
    async_wait0();
    __syncthreads();
    v16h af = frag_a(&sA[rt * 16][0], 40, lane);
#pragma unroll
    for (int t = 0; t < 2; ++t) {
      v16h bf = frag_bt(&sBt[ch + t * 16][0], 40, lane);
      acc[t] = wmma16(af, bf, acc[t]);
    }
    __syncthreads();
  }
  int nn = lane & 15, mb = (lane >> 4) * 8;
#pragma unroll
  for (int t = 0; t < 2; ++t)
#pragma unroll
    for (int j = 0; j < 8; ++j) {
      int tok = tok0 + rt * 16 + mb + j;
      if (tok < NT_) {
        float d = D[(long)(b * NT_ + tok) * H_ + h];
        CTX[(long)(b * NT_ + tok) * C_ + h * HD_ + ch + t * 16 + nn] =
            (half_t)(acc[t][j] / d);
      }
    }
}

__global__ void k_head(const float* __restrict__ X,
                       const float* __restrict__ g,
                       const float* __restrict__ beta,
                       const float* __restrict__ hw,
                       const float* __restrict__ hb,
                       float* __restrict__ out) {
  __shared__ float red[256];
  __shared__ float red2[256];
  int b = blockIdx.x, tid = threadIdx.x;
  const float* xr = X + (long)(b * NT_) * C_;
  float v0 = xr[tid], v1 = xr[tid + 256];
  red[tid] = v0 + v1;
  __syncthreads();
  for (int s = 128; s > 0; s >>= 1) { if (tid < s) red[tid] += red[tid + s]; __syncthreads(); }
  float mu = red[0] / (float)C_;
  __syncthreads();
  float d0 = v0 - mu, d1 = v1 - mu;
  red[tid] = d0 * d0 + d1 * d1;
  __syncthreads();
  for (int s = 128; s > 0; s >>= 1) { if (tid < s) red[tid] += red[tid + s]; __syncthreads(); }
  float rstd = rsqrtf(red[0] / (float)C_ + LN_EPS_);
  __syncthreads();
  float y0 = d0 * rstd * g[tid] + beta[tid];
  float y1 = d1 * rstd * g[tid + 256] + beta[tid + 256];
  red[tid]  = y0 * hw[tid * NC_ + 0] + y1 * hw[(tid + 256) * NC_ + 0];
  red2[tid] = y0 * hw[tid * NC_ + 1] + y1 * hw[(tid + 256) * NC_ + 1];
  __syncthreads();
  for (int s = 128; s > 0; s >>= 1) {
    if (tid < s) { red[tid] += red[tid + s]; red2[tid] += red2[tid + s]; }
    __syncthreads();
  }
  if (tid == 0) {
    out[b * NC_ + 0] = red[0] + hb[0];
    out[b * NC_ + 1] = red2[0] + hb[1];
  }
}

// =====================================================================
// Host orchestration
// =====================================================================
extern "C" void kernel_launch(void* const* d_in, const int* in_sizes, int n_in,
                              void* d_out, int out_size, void* d_ws, size_t ws_size,
                              hipStream_t stream) {
  const int*   ids   = (const int*)d_in[0];
  const int*   amask = (const int*)d_in[1];
  const float* tok   = (const float*)d_in[2];
  const float* cls   = (const float*)d_in[3];
  const float* pos   = (const float*)d_in[4];
  const float* ln1_g = (const float*)d_in[5];
  const float* ln1_b = (const float*)d_in[6];
  const float* qW    = (const float*)d_in[7];
  const float* qb    = (const float*)d_in[8];
  const float* kW    = (const float*)d_in[9];
  const float* kb    = (const float*)d_in[10];
  const float* vW    = (const float*)d_in[11];
  const float* vb    = (const float*)d_in[12];
  const float* oW    = (const float*)d_in[13];
  const float* ob    = (const float*)d_in[14];
  const float* proj  = (const float*)d_in[15];
  const float* ln2_g = (const float*)d_in[16];
  const float* ln2_b = (const float*)d_in[17];
  const float* fc1W  = (const float*)d_in[18];
  const float* fc1b  = (const float*)d_in[19];
  const float* fc2W  = (const float*)d_in[20];
  const float* fc2b  = (const float*)d_in[21];
  const float* lnf_g = (const float*)d_in[22];
  const float* lnf_b = (const float*)d_in[23];
  const float* headW = (const float*)d_in[24];
  const float* headb = (const float*)d_in[25];

  char* ws = (char*)d_ws;
  size_t off = 0;
  auto alloc = [&](size_t bytes) -> void* {
    void* p = ws + off;
    off += (bytes + 255) & ~(size_t)255;
    return p;
  };
  const long WCC  = (long)L_ * C_ * C_;
  const long WCH  = (long)L_ * C_ * HID_;
  const long WPRJ = (long)L_ * H_ * HD_ * M_;

  half_t* qWt  = (half_t*)alloc(WCC * 2);
  half_t* kWt  = (half_t*)alloc(WCC * 2);
  half_t* vWt  = (half_t*)alloc(WCC * 2);
  half_t* oWt  = (half_t*)alloc(WCC * 2);
  half_t* f1Wt = (half_t*)alloc(WCH * 2);
  half_t* f2Wt = (half_t*)alloc(WCH * 2);
  half_t* prjt = (half_t*)alloc(WPRJ * 2);

  float*  X    = (float*)alloc((long)R_ * C_ * 4);
  half_t* Yh   = (half_t*)alloc((long)R_ * C_ * 2);
  half_t* Qh   = (half_t*)alloc((long)R_ * C_ * 2);
  half_t* Kh   = (half_t*)alloc((long)R_ * C_ * 2);
  half_t* QFh  = (half_t*)alloc((long)R_ * HM_ * 2);
  half_t* KFt  = (half_t*)alloc((long)B_ * H_ * M_ * NTP_ * 2);
  half_t* Vt   = (half_t*)alloc((long)B_ * H_ * HD_ * NTP_ * 2);
  half_t* KVt  = (half_t*)alloc((long)B_ * H_ * M_ * HD_ * 2);
  float*  KS   = (float*)alloc((long)B_ * H_ * M_ * 4);
  float*  Dn   = (float*)alloc((long)R_ * H_ * 4);
  half_t* CTXh = (half_t*)alloc((long)R_ * C_ * 2);
  half_t* Hh   = (half_t*)alloc((long)R_ * HID_ * 2);
  (void)ws_size; (void)n_in; (void)in_sizes; (void)out_size;

  k_convert_T<<<dim3(C_/32,  C_/32,  L_),      256, 0, stream>>>(qW,  qWt,  C_,   C_);
  k_convert_T<<<dim3(C_/32,  C_/32,  L_),      256, 0, stream>>>(kW,  kWt,  C_,   C_);
  k_convert_T<<<dim3(C_/32,  C_/32,  L_),      256, 0, stream>>>(vW,  vWt,  C_,   C_);
  k_convert_T<<<dim3(C_/32,  C_/32,  L_),      256, 0, stream>>>(oW,  oWt,  C_,   C_);
  k_convert_T<<<dim3(C_/32,  HID_/32, L_),     256, 0, stream>>>(fc1W, f1Wt, C_,   HID_);
  k_convert_T<<<dim3(HID_/32, C_/32,  L_),     256, 0, stream>>>(fc2W, f2Wt, HID_, C_);
  k_convert_T<<<dim3(HD_/32, M_/32,  L_ * H_), 256, 0, stream>>>(proj, prjt, HD_,  M_);

  // one-time zero of padded-token buffers (pads stay zero across layers)
  k_zero_h<<<2048, 256, 0, stream>>>(KFt, (long)B_ * H_ * M_ * NTP_);
  k_zero_h<<<2048, 256, 0, stream>>>(Vt,  (long)B_ * H_ * HD_ * NTP_);

  k_embed<<<R_, 256, 0, stream>>>(ids, tok, cls, pos, X);

  const int RB = (R_ + 127) / 128;             // 129
  const dim3 gQKV(RB, C_ / 128);               // 129 x 4
  const dim3 gFC1(RB, HID_ / 128);             // 129 x 16
  const dim3 gFeat((R_ + 31) / 32, H_);        // 513 x 8
  const dim3 gCtx((NT_ + 63) / 64, B_ * H_);   // 33 x 64
  const dim3 gDen(R_, H_);

  for (int l = 0; l < L_; ++l) {
    const half_t* qw = qWt + (long)l * C_ * C_;
    const half_t* kw = kWt + (long)l * C_ * C_;
    const half_t* vw = vWt + (long)l * C_ * C_;
    const half_t* ow = oWt + (long)l * C_ * C_;
    const half_t* w1 = f1Wt + (long)l * C_ * HID_;
    const half_t* w2 = f2Wt + (long)l * HID_ * C_;
    const half_t* pj = prjt + (long)l * H_ * HD_ * M_;

    k_layernorm<<<R_, 256, 0, stream>>>(X, ln1_g + l * C_, ln1_b + l * C_, Yh);

    k_gemm<0><<<gQKV, 256, 0, stream>>>(Yh, C_, qw, qb + l * C_,
                                        nullptr, Qh, C_, R_, C_, SCALE_, nullptr);
    k_gemm<1><<<gQKV, 256, 0, stream>>>(Yh, C_, kw, kb + l * C_,
                                        nullptr, Kh, C_, R_, C_, 1.f, amask);
    k_gemm<4><<<gQKV, 256, 0, stream>>>(Yh, C_, vw, vb + l * C_,
                                        nullptr, Vt, 0, R_, C_, 1.f, amask);

    k_feature<false><<<gFeat, 256, 0, stream>>>(Qh, pj, QFh);
    k_feature<true ><<<gFeat, 256, 0, stream>>>(Kh, pj, KFt);

    k_kv<<<B_ * H_, 256, 0, stream>>>(KFt, Vt, KVt);
    k_ksum<<<(B_ * H_ * M_) / 256, 256, 0, stream>>>(KFt, KS);
    k_denom<<<gDen, M_, 0, stream>>>(QFh, KS, Dn);
    k_ctx<<<gCtx, 256, 0, stream>>>(QFh, KVt, Dn, CTXh);

    k_gemm<2><<<gQKV, 256, 0, stream>>>(CTXh, C_, ow, ob + l * C_,
                                        X, nullptr, C_, R_, C_, 1.f, nullptr);

    k_layernorm<<<R_, 256, 0, stream>>>(X, ln2_g + l * C_, ln2_b + l * C_, Yh);
    k_gemm<3><<<gFC1, 256, 0, stream>>>(Yh, C_, w1, fc1b + l * HID_,
                                        nullptr, Hh, HID_, R_, C_, 1.f, nullptr);
    k_gemm<2><<<gQKV, 256, 0, stream>>>(Hh, HID_, w2, fc2b + l * C_,
                                        X, nullptr, C_, R_, HID_, 1.f, nullptr);
  }

  k_head<<<B_, 256, 0, stream>>>(X, lnf_g, lnf_b, headW, headb, (float*)d_out);
}
